// ParallelDroplessMLP_2302102471525
// MI455X (gfx1250) — matrix operand order repaired and checked by reading, not compiled, not run on hardware
//
#include <hip/hip_runtime.h>
#include <math.h>

// ---------------------------------------------------------------------------
// Types for CDNA5 WMMA (wave32)
// ---------------------------------------------------------------------------
typedef __bf16 bf16_t;
typedef bf16_t v16bf __attribute__((ext_vector_type(16)));
typedef bf16_t v8bf  __attribute__((ext_vector_type(8)));
typedef bf16_t v4bf  __attribute__((ext_vector_type(4)));
typedef float  v8f   __attribute__((ext_vector_type(8)));
typedef int    v4i_t __attribute__((ext_vector_type(4)));

// Problem constants (fixed by the reference)
constexpr int NE   = 8;
constexpr int NTOK = 4096;       // S*B tokens
constexpr int D    = 1024;
constexpr int F    = 4096;
constexpr int NASG = NTOK * 2;   // 8192 assignment rows (top-2)

// GEMM tiling
constexpr int TM = 64;
constexpr int TN = 128;
constexpr int TK = 32;
constexpr int AST = 40;          // LDS A row stride (bf16), padded
constexpr int BST = 40;          // LDS B row stride (bf16), padded
constexpr int ASZ = TM * AST;    // one A buffer (elems)
constexpr int BSZ = TN * BST;    // one B buffer (elems)

// ---------------------------------------------------------------------------
// Workspace layout (bytes)
// ---------------------------------------------------------------------------
constexpr size_t OFF_XB   = 0;                                  // x bf16
constexpr size_t OFF_W1T  = OFF_XB  + (size_t)NTOK * D * 2;     // w1^T bf16 [e][F][D]
constexpr size_t OFF_W2T  = OFF_W1T + (size_t)NE * D * F * 2;   // w2^T bf16 [e][D][F]
constexpr size_t OFF_HB   = OFF_W2T + (size_t)NE * F * D * 2;   // h bf16 [8192][F]
constexpr size_t OFF_YB   = OFF_HB  + (size_t)NASG * F * 2;     // y f32 [8192][D]
constexpr size_t OFF_CNT  = OFF_YB  + (size_t)NASG * D * 4;
constexpr size_t OFF_OFFS = OFF_CNT  + 64;
constexpr size_t OFF_CUR  = OFF_OFFS + 64;
constexpr size_t OFF_BKT  = OFF_CUR  + 64;

// ---------------------------------------------------------------------------
// gfx1250 async global->LDS copies (ASYNCcnt path), with portable fallback
// ---------------------------------------------------------------------------
#if defined(__gfx1250__) && __has_builtin(__builtin_amdgcn_global_load_async_to_lds_b128)
#define USE_ASYNC 1
#else
#define USE_ASYNC 0
#endif

__device__ __forceinline__ void cp16(const bf16_t* g, bf16_t* l) {
#if USE_ASYNC
    __builtin_amdgcn_global_load_async_to_lds_b128(
        (v4i_t*)(void*)const_cast<bf16_t*>(g),
        (v4i_t*)(void*)l, 0, 0);
#else
    *reinterpret_cast<v8bf*>(l) = *reinterpret_cast<const v8bf*>(g);
#endif
}

__device__ __forceinline__ void cp_wait3() {   // leave next tile's 3 copies in flight
#if USE_ASYNC
#if __has_builtin(__builtin_amdgcn_s_wait_asynccnt)
    __builtin_amdgcn_s_wait_asynccnt(3);
#else
    asm volatile("s_wait_asynccnt 3" ::: "memory");
#endif
#endif
}
__device__ __forceinline__ void cp_wait0() {
#if USE_ASYNC
#if __has_builtin(__builtin_amdgcn_s_wait_asynccnt)
    __builtin_amdgcn_s_wait_asynccnt(0);
#else
    asm volatile("s_wait_asynccnt 0" ::: "memory");
#endif
#endif
}

// ---------------------------------------------------------------------------
// Small helpers
// ---------------------------------------------------------------------------
__global__ void k_zero_small(int* counts, int* cursor) {
    int t = threadIdx.x;
    if (t < NE) { counts[t] = 0; cursor[t] = 0; }
}

__global__ void k_cvt_bf16(const float* __restrict__ src,
                           bf16_t* __restrict__ dst, int n4) {
    int i = blockIdx.x * blockDim.x + threadIdx.x;
    if (i >= n4) return;
    const float4 v = reinterpret_cast<const float4*>(src)[i];
    v4bf o;
    o[0] = (bf16_t)v.x; o[1] = (bf16_t)v.y; o[2] = (bf16_t)v.z; o[3] = (bf16_t)v.w;
    reinterpret_cast<v4bf*>(dst)[i] = o;
}

// Convert + transpose: src f32 [E][K][Nn] -> dst bf16 [E][Nn][K] (32x32 tiles)
// grid: (K/32, Nn/32, E), block 256
__global__ __launch_bounds__(256)
void k_cvt_tr(const float* __restrict__ src, bf16_t* __restrict__ dst,
              int K, int Nn) {
    __shared__ bf16_t T[32][36];
    const int e = blockIdx.z;
    const size_t eb = (size_t)e * K * Nn;
    const int k0 = blockIdx.x * 32, nb0 = blockIdx.y * 32;
    const int t = threadIdx.x;
    {
        int kr = t >> 3, c4 = t & 7;   // 32 rows x 8 float4 chunks
        float4 v = *reinterpret_cast<const float4*>(
            src + eb + (size_t)(k0 + kr) * Nn + nb0 + c4 * 4);
        T[kr][c4 * 4 + 0] = (bf16_t)v.x;
        T[kr][c4 * 4 + 1] = (bf16_t)v.y;
        T[kr][c4 * 4 + 2] = (bf16_t)v.z;
        T[kr][c4 * 4 + 3] = (bf16_t)v.w;
    }
    __syncthreads();
    {
        int nr = t >> 3, ch = t & 7;   // 32 out-rows x 8 chunks of 4 k
        v4bf o;
        #pragma unroll
        for (int j = 0; j < 4; ++j) o[j] = T[ch * 4 + j][nr];
        *reinterpret_cast<v4bf*>(dst + eb + (size_t)(nb0 + nr) * K + k0 + ch * 4) = o;
    }
}

__global__ void k_count(const int* __restrict__ idx, int* counts) {
    int i = blockIdx.x * blockDim.x + threadIdx.x;
    if (i >= NASG) return;
    atomicAdd(&counts[idx[i] & 7], 1);
}

__global__ void k_scan(const int* __restrict__ counts, int* offsets) {
    if (threadIdx.x == 0) {
        int acc = 0;
        for (int e = 0; e < NE; ++e) { offsets[e] = acc; acc += counts[e]; }
        offsets[NE] = acc;
    }
}

__global__ void k_scatter(const int* __restrict__ idx,
                          const int* __restrict__ offsets,
                          int* cursor, int* bucket) {
    int i = blockIdx.x * blockDim.x + threadIdx.x;
    if (i >= NASG) return;
    int e = idx[i] & 7;
    int p = atomicAdd(&cursor[e], 1);
    bucket[offsets[e] + p] = i;      // assignment id = token*2 + slot
}

// ---------------------------------------------------------------------------
// WMMA fragment helpers (ISA 7.12.2 layouts)
// ---------------------------------------------------------------------------
__device__ __forceinline__ v16bf frag_a(const bf16_t* As, int wm, int lane) {
    const bool lo = lane < 16;
    const int arow = wm * 16 + (lane & 15);
    const int aoff = lo ? 0 : 8;
    v8bf a0 = *reinterpret_cast<const v8bf*>(&As[arow * AST + aoff]);
    v8bf a1 = *reinterpret_cast<const v8bf*>(&As[arow * AST + 16 + aoff]);
    return __builtin_shufflevector(a0, a1, 0,1,2,3,4,5,6,7,8,9,10,11,12,13,14,15);
}
__device__ __forceinline__ v16bf frag_b(const bf16_t* Bs, int col, int lane) {
    const int boff = (lane < 16) ? 0 : 8;
    const int bcol = col + (lane & 15);
    v8bf b0 = *reinterpret_cast<const v8bf*>(&Bs[bcol * BST + boff]);
    v8bf b1 = *reinterpret_cast<const v8bf*>(&Bs[bcol * BST + 16 + boff]);
    return __builtin_shufflevector(b0, b1, 0,1,2,3,4,5,6,7,8,9,10,11,12,13,14,15);
}

// ---------------------------------------------------------------------------
// GEMM1: h = GeLU(x_gather @ w1[e]); rows = expert bucket, K = D, Nout = F.
// grid: (NASG/TM, F/TN, NE), block 256 (8 waves). Double-buffered async LDS.
// ---------------------------------------------------------------------------
__global__ __launch_bounds__(256)
void k_gemm1(const bf16_t* __restrict__ xb, const bf16_t* __restrict__ w1t,
             bf16_t* __restrict__ hb, const int* __restrict__ counts,
             const int* __restrict__ offsets, const int* __restrict__ bucket) {
    const int e   = blockIdx.z;
    const int cnt = counts[e];
    const int mt  = blockIdx.x;
    if (mt * TM >= cnt) return;
    const int base = offsets[e];
    const int n0   = blockIdx.y * TN;

    __shared__ bf16_t As[2 * ASZ];
    __shared__ bf16_t Bs[2 * BSZ];
    __shared__ float  Es[TM][TN + 4];

    const int tid  = threadIdx.x;
    const int wave = tid >> 5;
    const int lane = tid & 31;
    const int wm   = wave & 3;
    const int wn   = wave >> 2;
    const bool lo  = lane < 16;
    const int l16  = lane & 15;

    // loop-invariant staging addresses (token gather resolved ONCE per thread)
    const size_t wb = (size_t)e * D * F;       // w1t: [e][F rows][D cols]
    const int ar = tid >> 2, ac = tid & 3;     // A: 64 rows x 4 chunks
    const int agr = mt * TM + ar;
    const int tok = ((agr < cnt) ? bucket[base + agr] : bucket[base]) >> 1;
    const bf16_t* aSrc = xb + (size_t)tok * D + ac * 8;
    const int aOff = ar * AST + ac * 8;

    const int bn = tid >> 2, bch = tid & 3;    // B: 128 rows x 4 chunks, 2 passes
    const bf16_t* bSrc0 = w1t + wb + (size_t)(n0 + bn) * D + bch * 8;
    const bf16_t* bSrc1 = bSrc0 + (size_t)64 * D;
    const int bOff0 = bn * BST + bch * 8;
    const int bOff1 = bOff0 + 64 * BST;

    auto issue = [&](int kb, bf16_t* Ad, bf16_t* Bd) {
        cp16(aSrc + kb, Ad + aOff);
        cp16(bSrc0 + kb, Bd + bOff0);
        cp16(bSrc1 + kb, Bd + bOff1);
        if (kb + 2 * TK < D)
            __builtin_prefetch(bSrc0 + kb + 2 * TK, 0, 0);
    };

    v8f acc[4] = {};
    issue(0, As, Bs);
    for (int kb = 0, buf = 0; kb < D; kb += TK, buf ^= 1) {
        bf16_t* Ac = As + buf * ASZ;
        bf16_t* Bc = Bs + buf * BSZ;
        if (kb + TK < D) {
            issue(kb + TK, As + (buf ^ 1) * ASZ, Bs + (buf ^ 1) * BSZ);
            cp_wait3();
        } else {
            cp_wait0();
        }
        __syncthreads();

        v16bf af = frag_a(Ac, wm, lane);
        #pragma unroll
        for (int c = 0; c < 4; ++c) {
            v16bf bf = frag_b(Bc, wn * 64 + c * 16, lane);
            acc[c] = __builtin_amdgcn_wmma_f32_16x16x32_bf16(
                false, af, false, bf, (short)0, acc[c], false, false);
        }
        __syncthreads();
    }

    // epilogue via LDS -> coalesced bf16 stores with exact GeLU
    #pragma unroll
    for (int c = 0; c < 4; ++c)
        #pragma unroll
        for (int r = 0; r < 8; ++r)
            Es[wm * 16 + r + (lo ? 0 : 8)][wn * 64 + c * 16 + l16] = acc[c][r];
    __syncthreads();
    #pragma unroll
    for (int p = 0; p < 4; ++p) {
        int u = tid + p * 256;             // 64 rows x 16 chunks of 8 cols
        int r = u >> 4, ch = u & 15;
        int gr = mt * TM + r;
        if (gr < cnt) {
            v8bf o;
            #pragma unroll
            for (int j = 0; j < 8; ++j) {
                float v = Es[r][ch * 8 + j];
                o[j] = (bf16_t)(0.5f * v * (1.0f + erff(v * 0.70710678118654752f)));
            }
            *reinterpret_cast<v8bf*>(hb + (size_t)(base + gr) * F + n0 + ch * 8) = o;
        }
    }
}

// ---------------------------------------------------------------------------
// GEMM2: y = h @ w2[e]; rows contiguous in hb, K = F, Nout = D.
// grid: (NASG/TM, D/TN, NE)
// ---------------------------------------------------------------------------
__global__ __launch_bounds__(256)
void k_gemm2(const bf16_t* __restrict__ hb, const bf16_t* __restrict__ w2t,
             float* __restrict__ yb, const int* __restrict__ counts,
             const int* __restrict__ offsets, const int* __restrict__ bucket) {
    const int e   = blockIdx.z;
    const int cnt = counts[e];
    const int mt  = blockIdx.x;
    if (mt * TM >= cnt) return;
    const int base = offsets[e];
    const int n0   = blockIdx.y * TN;

    __shared__ bf16_t As[2 * ASZ];
    __shared__ bf16_t Bs[2 * BSZ];
    __shared__ float  Es[TM][TN + 4];

    const int tid  = threadIdx.x;
    const int wave = tid >> 5;
    const int lane = tid & 31;
    const int wm   = wave & 3;
    const int wn   = wave >> 2;
    const bool lo  = lane < 16;
    const int l16  = lane & 15;

    const size_t wb = (size_t)e * F * D;       // w2t: [e][D rows][F cols]
    const int ar = tid >> 2, ac = tid & 3;
    const int agr = mt * TM + ar;
    const size_t hrow = (agr < cnt) ? (size_t)(base + agr) : (size_t)base;
    const bf16_t* aSrc = hb + hrow * F + ac * 8;
    const int aOff = ar * AST + ac * 8;

    const int bn = tid >> 2, bch = tid & 3;
    const bf16_t* bSrc0 = w2t + wb + (size_t)(n0 + bn) * F + bch * 8;
    const bf16_t* bSrc1 = bSrc0 + (size_t)64 * F;
    const int bOff0 = bn * BST + bch * 8;
    const int bOff1 = bOff0 + 64 * BST;

    auto issue = [&](int kb, bf16_t* Ad, bf16_t* Bd) {
        cp16(aSrc + kb, Ad + aOff);
        cp16(bSrc0 + kb, Bd + bOff0);
        cp16(bSrc1 + kb, Bd + bOff1);
        if (kb + 2 * TK < F)
            __builtin_prefetch(bSrc0 + kb + 2 * TK, 0, 0);
    };

    v8f acc[4] = {};
    issue(0, As, Bs);
    for (int kb = 0, buf = 0; kb < F; kb += TK, buf ^= 1) {
        bf16_t* Ac = As + buf * ASZ;
        bf16_t* Bc = Bs + buf * BSZ;
        if (kb + TK < F) {
            issue(kb + TK, As + (buf ^ 1) * ASZ, Bs + (buf ^ 1) * BSZ);
            cp_wait3();
        } else {
            cp_wait0();
        }
        __syncthreads();

        v16bf af = frag_a(Ac, wm, lane);
        #pragma unroll
        for (int c = 0; c < 4; ++c) {
            v16bf bf = frag_b(Bc, wn * 64 + c * 16, lane);
            acc[c] = __builtin_amdgcn_wmma_f32_16x16x32_bf16(
                false, af, false, bf, (short)0, acc[c], false, false);
        }
        __syncthreads();
    }

    #pragma unroll
    for (int c = 0; c < 4; ++c)
        #pragma unroll
        for (int r = 0; r < 8; ++r)
            Es[wm * 16 + r + (lo ? 0 : 8)][wn * 64 + c * 16 + l16] = acc[c][r];
    __syncthreads();
    #pragma unroll
    for (int p = 0; p < 8; ++p) {
        int u = tid + p * 256;             // 64 rows x 32 float4 chunks
        int r = u >> 5, ch = u & 31;
        int gr = mt * TM + r;
        if (gr < cnt) {
            int a = bucket[base + gr];     // assignment id for deterministic scatter
            float4 o = *reinterpret_cast<const float4*>(&Es[r][ch * 4]);
            *reinterpret_cast<float4*>(yb + (size_t)a * D + n0 + ch * 4) = o;
        }
    }
}

// ---------------------------------------------------------------------------
// Combine: out[n] = w[n,0]*y[2n] + w[n,1]*y[2n+1]
// ---------------------------------------------------------------------------
__global__ void k_combine(const float* __restrict__ yb,
                          const float* __restrict__ ew,
                          float* __restrict__ out) {
    int i = blockIdx.x * blockDim.x + threadIdx.x;
    if (i >= NTOK * (D / 4)) return;
    int n  = i / (D / 4);
    int c4 = i % (D / 4);
    const float4 y0 = reinterpret_cast<const float4*>(yb + (size_t)(2 * n)     * D)[c4];
    const float4 y1 = reinterpret_cast<const float4*>(yb + (size_t)(2 * n + 1) * D)[c4];
    float w0 = ew[2 * n], w1 = ew[2 * n + 1];
    float4 o;
    o.x = w0 * y0.x + w1 * y1.x;
    o.y = w0 * y0.y + w1 * y1.y;
    o.z = w0 * y0.z + w1 * y1.z;
    o.w = w0 * y0.w + w1 * y1.w;
    reinterpret_cast<float4*>(out + (size_t)n * D)[c4] = o;
}

// ---------------------------------------------------------------------------
// Launch
// ---------------------------------------------------------------------------
extern "C" void kernel_launch(void* const* d_in, const int* in_sizes, int n_in,
                              void* d_out, int out_size, void* d_ws, size_t ws_size,
                              hipStream_t stream) {
    (void)in_sizes; (void)n_in; (void)out_size; (void)ws_size;
    const float* x   = (const float*)d_in[0];
    const float* ew  = (const float*)d_in[1];
    const int*   idx = (const int*)d_in[2];
    const float* w1  = (const float*)d_in[4];
    const float* w2  = (const float*)d_in[5];
    float* out = (float*)d_out;

    char* ws = (char*)d_ws;
    bf16_t* xb   = (bf16_t*)(ws + OFF_XB);
    bf16_t* w1t  = (bf16_t*)(ws + OFF_W1T);
    bf16_t* w2t  = (bf16_t*)(ws + OFF_W2T);
    bf16_t* hb   = (bf16_t*)(ws + OFF_HB);
    float*  yb   = (float*)(ws + OFF_YB);
    int* counts  = (int*)(ws + OFF_CNT);
    int* offsets = (int*)(ws + OFF_OFFS);
    int* cursor  = (int*)(ws + OFF_CUR);
    int* bucket  = (int*)(ws + OFF_BKT);

    hipLaunchKernelGGL(k_zero_small, dim3(1), dim3(32), 0, stream, counts, cursor);

    // fp32 -> bf16 (x linear; weights converted + transposed to [e][n][k])
    {
        int n4 = NTOK * D / 4;
        hipLaunchKernelGGL(k_cvt_bf16, dim3((n4 + 255) / 256), dim3(256), 0, stream, x, xb, n4);
    }
    hipLaunchKernelGGL(k_cvt_tr, dim3(D / 32, F / 32, NE), dim3(256), 0, stream, w1, w1t, D, F);
    hipLaunchKernelGGL(k_cvt_tr, dim3(F / 32, D / 32, NE), dim3(256), 0, stream, w2, w2t, F, D);

    // routing
    hipLaunchKernelGGL(k_count,   dim3(NASG / 256), dim3(256), 0, stream, idx, counts);
    hipLaunchKernelGGL(k_scan,    dim3(1), dim3(1), 0, stream, counts, offsets);
    hipLaunchKernelGGL(k_scatter, dim3(NASG / 256), dim3(256), 0, stream, idx, offsets, cursor, bucket);

    // grouped MLP
    hipLaunchKernelGGL(k_gemm1, dim3(NASG / TM, F / TN, NE), dim3(256), 0, stream,
                       xb, w1t, hb, counts, offsets, bucket);
    hipLaunchKernelGGL(k_gemm2, dim3(NASG / TM, D / TN, NE), dim3(256), 0, stream,
                       hb, w2t, yb, counts, offsets, bucket);

    // weighted combine
    {
        int n = NTOK * (D / 4);
        hipLaunchKernelGGL(k_combine, dim3((n + 255) / 256), dim3(256), 0, stream, yb, ew, out);
    }
}
